// FreqPerceiver_40252433498788
// MI455X (gfx1250) — compile-verified, matrix operand order checked by reading, not run on hardware
//
#include <hip/hip_runtime.h>
#include <hip/hip_bf16.h>

typedef __attribute__((ext_vector_type(16))) _Float16 v16h;
typedef __attribute__((ext_vector_type(8)))  float    v8f;

#define Bq 8
#define Hq 16
#define Wq 512
#define Dq 512
#define NHEADS 8
#define DHEAD 64
#define NTOT (Bq*Wq)          // 4096
#define KCH 16                // K chunks of 32 (K=512)
#define NT_PER_HEAD 4         // 4 x 16-col tiles per head

// CDNA5 16-bit operand K striping within a 32-wide K chunk (ISA 7.12.2):
// lanes 0-15:  half i<8 -> K=i,    i>=8 -> K=i+8   (covers 0-7, 16-23)
// lanes 16-31: half i<8 -> K=i+8,  i>=8 -> K=i+16  (covers 8-15, 24-31)
__device__ __forceinline__ int kstripe(int lane, int i) {
    int kb = (i < 8) ? i : i + 8;
    if (lane >= 16) kb += 8;
    return kb;
}

// ---------------------------------------------------------------------------
// Kernel 0: pack kp_w / vp_w (fp32 [512][512], row = out col) into f16
// WMMA-B operand layout: [nt(32)][kc(16)][lane(32)][16 halfs] (32B/lane chunk)
// ---------------------------------------------------------------------------
__global__ __launch_bounds__(256) void prep_w_kernel(
    const float* __restrict__ kp_w, const float* __restrict__ vp_w,
    _Float16* __restrict__ Bk, _Float16* __restrict__ Bv)
{
    int tid  = blockIdx.x * 256 + threadIdx.x;   // 64 blocks * 256 = 16384
    int lane = tid & 31;
    int pair = tid >> 5;                         // 0..511  (nt*16 + kc)
    int nt   = pair >> 4;
    int kc   = pair & 15;
    int col  = nt * 16 + (lane & 15);
    __align__(32) _Float16 tk[16], tv[16];
    #pragma unroll
    for (int i = 0; i < 16; ++i) {
        int k = kc * 32 + kstripe(lane, i);
        tk[i] = (_Float16)kp_w[(size_t)col * Dq + k];
        tv[i] = (_Float16)vp_w[(size_t)col * Dq + k];
    }
    size_t off = ((size_t)pair * 32 + lane) * 16;
    *(v16h*)(Bk + off) = *(v16h*)tk;
    *(v16h*)(Bv + off) = *(v16h*)tv;
}

// ---------------------------------------------------------------------------
// Kernel 1: x = swin + pe  (one block per n = b*W + w)
//   - memory_k[n,d] = sum_h x[n,h,d]*k_w[d,h] + k_b[d]     (fp32)
//   - pack x slab [16][512] into f16 WMMA-A layout: [n][kc][lane][16 halfs]
// ---------------------------------------------------------------------------
__global__ __launch_bounds__(256) void prep_x_kernel(
    const float* __restrict__ swin, const float* __restrict__ pe,
    const float* __restrict__ k_w, const float* __restrict__ k_b,
    _Float16* __restrict__ Apk, float* __restrict__ mem_k)
{
    __shared__ float xs[Hq][Dq];          // 32 KB
    int n = blockIdx.x;                   // 0..4095
    int b = n >> 9, w = n & 511;
    int t = threadIdx.x;                  // 0..255

    #pragma unroll
    for (int h = 0; h < Hq; ++h) {
        int d = t * 2;
        const float2 s = *(const float2*)(swin + (((size_t)(b*Hq + h)*Wq + w)*Dq + d));
        const float2 p = *(const float2*)(pe   + (((size_t)h*Wq + w)*Dq + d));
        xs[h][d]   = s.x + p.x;
        xs[h][d+1] = s.y + p.y;
    }
    __syncthreads();

    // depthwise conv over H (fp32 accuracy)
    #pragma unroll
    for (int dd = 0; dd < 2; ++dd) {
        int d = t + dd * 256;
        float acc = 0.f;
        #pragma unroll
        for (int h = 0; h < Hq; ++h) acc += xs[h][d] * k_w[d * Hq + h];
        mem_k[(size_t)n * Dq + d] = acc + k_b[d];
    }

    // pack A operand chunks: p = kc*32 + lane, 2 chunks per thread
    #pragma unroll
    for (int pp = 0; pp < 2; ++pp) {
        int p    = t * 2 + pp;
        int kc   = p >> 5, lane = p & 31;
        int m    = lane & 15;             // A row (= h)
        __align__(32) _Float16 tmp[16];
        #pragma unroll
        for (int i = 0; i < 16; ++i)
            tmp[i] = (_Float16)xs[m][kc * 32 + kstripe(lane, i)];
        *(v16h*)(Apk + (((size_t)n * KCH + kc) * 32 + lane) * 16) = *(v16h*)tmp;
    }
}

// ---------------------------------------------------------------------------
// Kernel 2: fused k/v projection (WMMA) + per-head cross-attention.
// grid = (head 0..7, chunk 0..127); block = 128 threads = 4 waves.
// Each workgroup stages its head's 128KB of f16 weights in LDS once, then
// each wave processes 8 n's: k,v tiles via v_wmma_f32_16x16x32_f16,
// 16-bin softmax, attention-weighted v -> pitch_out (pre-LN) to scratch.
// ---------------------------------------------------------------------------
__global__ __launch_bounds__(128) void attn_gemm_kernel(
    const _Float16* __restrict__ Apk,
    const _Float16* __restrict__ Bk, const _Float16* __restrict__ Bv,
    const float* __restrict__ kp_b, const float* __restrict__ vp_b,
    const float* __restrict__ pq, float* __restrict__ po)
{
    extern __shared__ __align__(32) char smem[];
    _Float16* Bks   = (_Float16*)(smem);            // 65536 B
    _Float16* Bvs   = (_Float16*)(smem + 65536);    // 65536 B
    float*    kS    = (float*)(smem + 131072);      // 4 waves * 16 * 64 * 4 = 16384
    float*    vS    = (float*)(smem + 147456);      // 16384
    float*    attnS = (float*)(smem + 163840);      // 4 * 16 * 4 = 256

    const int head  = blockIdx.x;     // 0..7
    const int chunk = blockIdx.y;     // 0..127
    const int t     = threadIdx.x;    // 0..127
    const int wave  = t >> 5;
    const int lane  = t & 31;

    // stage this head's weight slice (contiguous 32768 halfs per matrix)
    const uint4* sk = (const uint4*)(Bk + (size_t)head * 32768);
    const uint4* sv = (const uint4*)(Bv + (size_t)head * 32768);
    uint4* dk = (uint4*)Bks;
    uint4* dv = (uint4*)Bvs;
    for (int i = t; i < 4096; i += 128) { dk[i] = sk[i]; dv[i] = sv[i]; }
    __syncthreads();

    const v16h* Av  = (const v16h*)Apk;
    const v16h* Bkv = (const v16h*)Bks;
    const v16h* Bvv = (const v16h*)Bvs;
    float* kW = kS + wave * (16 * 64);
    float* vW = vS + wave * (16 * 64);
    float* aW = attnS + wave * 16;

    for (int it = 0; it < 8; ++it) {
        int n = chunk * 32 + it * 4 + wave;

        // A operand: 16 K-chunks, one 32B contiguous load per lane per chunk
        v16h a[KCH];
        #pragma unroll
        for (int kc = 0; kc < KCH; ++kc)
            a[kc] = Av[((size_t)n * KCH + kc) * 32 + lane];
        if (it + 1 < 8)
            __builtin_prefetch(Apk + ((size_t)(n + 4) * KCH) * 32 * 16, 0, 1);

        const int mbase = (lane < 16) ? 0 : 8;
        const int lcol  = lane & 15;
        #pragma unroll
        for (int nt = 0; nt < NT_PER_HEAD; ++nt) {
            v8f ck = {0.f, 0.f, 0.f, 0.f, 0.f, 0.f, 0.f, 0.f};
            v8f cv = {0.f, 0.f, 0.f, 0.f, 0.f, 0.f, 0.f, 0.f};
            #pragma unroll
            for (int kc = 0; kc < KCH; ++kc) {
                v16h bk = Bkv[(nt * KCH + kc) * 32 + lane];
                v16h bv = Bvv[(nt * KCH + kc) * 32 + lane];
                ck = __builtin_amdgcn_wmma_f32_16x16x32_f16(
                        false, a[kc], false, bk, (short)0, ck, false, false);
                cv = __builtin_amdgcn_wmma_f32_16x16x32_f16(
                        false, a[kc], false, bv, (short)0, cv, false, false);
            }
            int col  = nt * 16 + lcol;                 // 0..63 within head
            float bk_add = kp_b[head * DHEAD + col];
            float bv_add = vp_b[head * DHEAD + col];
            #pragma unroll
            for (int r = 0; r < 8; ++r) {
                kW[(mbase + r) * 64 + col] = ck[r] + bk_add;
                vW[(mbase + r) * 64 + col] = cv[r] + bv_add;
            }
        }
        __syncthreads();

        // scores over H=16 freq bins: lane (mod 16) owns one bin
        int f = lane & 15;
        float sc = 0.f;
        #pragma unroll
        for (int c = 0; c < DHEAD; ++c)
            sc += kW[f * 64 + c] * pq[head * DHEAD + c];
        sc *= 0.125f;                                  // 1/sqrt(64)
        float mx = sc;
        #pragma unroll
        for (int d = 8; d >= 1; d >>= 1) mx = fmaxf(mx, __shfl_xor(mx, d, 32));
        float e = __expf(sc - mx);
        float sm = e;
        #pragma unroll
        for (int d = 8; d >= 1; d >>= 1) sm += __shfl_xor(sm, d, 32);
        if (lane < 16) aW[f] = e / sm;
        __syncthreads();

        // pitch_out[c] = sum_f attn[f] * v[f,c]  (2 cols per lane)
        #pragma unroll
        for (int cc = 0; cc < 2; ++cc) {
            int c = lane + cc * 32;
            float acc = 0.f;
            #pragma unroll
            for (int f2 = 0; f2 < 16; ++f2) acc += aW[f2] * vW[f2 * 64 + c];
            po[(size_t)n * Dq + head * DHEAD + c] = acc;
        }
        __syncthreads();
    }
}

// ---------------------------------------------------------------------------
// Kernel 3: LayerNorm over D=512 per n -> memory_v
// ---------------------------------------------------------------------------
__global__ __launch_bounds__(256) void ln_kernel(
    const float* __restrict__ po, const float* __restrict__ g,
    const float* __restrict__ beta, float* __restrict__ mv)
{
    __shared__ float rs[8], rss[8], sh_mean, sh_inv;
    int n = blockIdx.x;
    int t = threadIdx.x;                  // 256 threads, 2 elems each
    float2 x = *(const float2*)(po + (size_t)n * Dq + t * 2);
    float s  = x.x + x.y;
    float ss = x.x * x.x + x.y * x.y;
    #pragma unroll
    for (int d = 16; d >= 1; d >>= 1) {
        s  += __shfl_xor(s,  d, 32);
        ss += __shfl_xor(ss, d, 32);
    }
    if ((t & 31) == 0) { rs[t >> 5] = s; rss[t >> 5] = ss; }
    __syncthreads();
    if (t == 0) {
        float ts = 0.f, tss = 0.f;
        #pragma unroll
        for (int i = 0; i < 8; ++i) { ts += rs[i]; tss += rss[i]; }
        float mean = ts * (1.0f / Dq);
        float var  = tss * (1.0f / Dq) - mean * mean;
        sh_mean = mean;
        sh_inv  = rsqrtf(var + 1e-5f);
    }
    __syncthreads();
    float mean = sh_mean, inv = sh_inv;
    int d = t * 2;
    float2 o;
    o.x = (x.x - mean) * inv * g[d]     + beta[d];
    o.y = (x.y - mean) * inv * g[d + 1] + beta[d + 1];
    *(float2*)(mv + (size_t)n * Dq + d) = o;
}

// ---------------------------------------------------------------------------
extern "C" void kernel_launch(void* const* d_in, const int* in_sizes, int n_in,
                              void* d_out, int out_size, void* d_ws, size_t ws_size,
                              hipStream_t stream) {
    const float* swin = (const float*)d_in[0];
    const float* pe   = (const float*)d_in[1];
    const float* pq   = (const float*)d_in[2];
    const float* k_w  = (const float*)d_in[3];
    const float* k_b  = (const float*)d_in[4];
    const float* kp_w = (const float*)d_in[5];
    const float* kp_b = (const float*)d_in[6];
    const float* vp_w = (const float*)d_in[7];
    const float* vp_b = (const float*)d_in[8];
    const float* ln_g = (const float*)d_in[9];
    const float* ln_b = (const float*)d_in[10];

    float* mem_k = (float*)d_out;                       // [B,W,D]
    float* mem_v = (float*)d_out + (size_t)NTOT * Dq;   // [B,W,D]

    char* ws = (char*)d_ws;
    _Float16* Apk = (_Float16*)(ws);                         // 64 MB
    _Float16* Bk  = (_Float16*)(ws + 67108864);              // 512 KB
    _Float16* Bv  = (_Float16*)(ws + 67108864 + 524288);     // 512 KB
    float*    po  = (float*)   (ws + 67108864 + 1048576);    // 8 MB

    prep_w_kernel<<<64, 256, 0, stream>>>(kp_w, vp_w, Bk, Bv);
    prep_x_kernel<<<NTOT, 256, 0, stream>>>(swin, pe, k_w, k_b, Apk, mem_k);
    attn_gemm_kernel<<<dim3(NHEADS, 128), 128, 164096, stream>>>(
        Apk, Bk, Bv, kp_b, vp_b, pq, po);
    ln_kernel<<<NTOT, 256, 0, stream>>>(po, ln_g, ln_b, mem_v);
}
